// E_GCL_11751030522785
// MI455X (gfx1250) — matrix operand order repaired and checked by reading, not compiled, not run on hardware
//
#include <hip/hip_runtime.h>
#include <hip/hip_bf16.h>

typedef _Float16 v16h __attribute__((ext_vector_type(16)));
typedef _Float16 v8h  __attribute__((ext_vector_type(8)));
typedef float    v8f  __attribute__((ext_vector_type(8)));

#define WE1_STRIDE 168   // padded from 160 (layer-1 K padded 131->160)
#define W64_STRIDE 72    // padded from 64
#define WN1_STRIDE 136   // padded from 128

__device__ __forceinline__ v8f wmma_f16(v16h a, v16h b, v8f c) {
    return __builtin_amdgcn_wmma_f32_16x16x32_f16(
        /*neg_a=*/false, a, /*neg_b=*/false, b,
        /*c_mod=*/(short)0, c, /*reuse_a=*/false, /*reuse_b=*/false);
}

__device__ __forceinline__ float silu(float x) {
    return x * (1.0f / (1.0f + __expf(-x)));
}

// A-fragment (16x32 f16): lane m=L%16, ks = chunk*32 + (L/16)*16 + 0..15
// (contiguous, matching b_frag), base = chunk*32 + (L/16)*8. rowbase points at
// this lane's contiguous fp32 row.
__device__ __forceinline__ v16h a_frag(const float* rowbase, int base) {
    v16h a;
#pragma unroll
    for (int j = 0; j < 8; ++j) {
        a[j]     = (_Float16)rowbase[base + (base & 8) + j];
        a[8 + j] = (_Float16)rowbase[base + (base & 8) + 8 + j];
    }
    return a;
}

// B-fragment (32x16 f16): lane n=L%16, K-half selected by L/16; 16 contiguous
// f16 from the transposed-weight LDS row.
__device__ __forceinline__ v16h b_frag(const _Float16* wt_row, int kbase) {
    v16h b;
#pragma unroll
    for (int j = 0; j < 16; ++j) b[j] = wt_row[kbase + j];
    return b;
}

// ---------------- Edge kernel: 4 waves/block, 16 edges/wave ----------------
__global__ __launch_bounds__(128) void egcl_edge_kernel(
    const float* __restrict__ h, const float* __restrict__ coord,
    const int* __restrict__ edge_index, const float* __restrict__ edge_attr,
    const float* __restrict__ W_e1, const float* __restrict__ b_e1,
    const float* __restrict__ W_e2, const float* __restrict__ b_e2,
    const float* __restrict__ W_c1, const float* __restrict__ b_c1,
    const float* __restrict__ W_c2,
    float* __restrict__ agg_h, float* __restrict__ agg_c,
    float* __restrict__ cnt, int E_)
{
    __shared__ _Float16 wt_e1[64 * WE1_STRIDE];
    __shared__ _Float16 wt_e2[64 * W64_STRIDE];
    __shared__ _Float16 wt_c1[64 * W64_STRIDE];
    __shared__ float act[4][16 * 68];
    __shared__ float cdiff_s[4][16 * 3];
    __shared__ float radial_s[4][16];
    __shared__ int   rows_s[4][16];
    __shared__ int   cols_s[4][16];

    const int tid = threadIdx.x;
    // Cooperative transposed f16 weight load (zero-pad K beyond 131).
    for (int i = tid; i < 64 * 160; i += 128) {
        int n = i / 160, k = i % 160;
        wt_e1[n * WE1_STRIDE + k] = (_Float16)((k < 131) ? W_e1[k * 64 + n] : 0.0f);
    }
    for (int i = tid; i < 64 * 64; i += 128) {
        int n = i / 64, k = i % 64;
        wt_e2[n * W64_STRIDE + k] = (_Float16)W_e2[k * 64 + n];
        wt_c1[n * W64_STRIDE + k] = (_Float16)W_c1[k * 64 + n];
    }
    __syncthreads();

    const int wv = tid >> 5, lane = tid & 31;
    const int m16 = lane & 15, hi = lane >> 4;
    const long e0 = (long)(blockIdx.x * 4 + wv) * 16;
    if (e0 >= (long)E_) return;

    // Per-edge prep (lanes 0..15): indices, coord_diff, radial -> LDS.
    if (lane < 16) {
        int e = (int)e0 + lane;
        int r = edge_index[e], c = edge_index[E_ + e];
        rows_s[wv][lane] = r; cols_s[wv][lane] = c;
        float dx = coord[r * 3 + 0] - coord[c * 3 + 0];
        float dy = coord[r * 3 + 1] - coord[c * 3 + 1];
        float dz = coord[r * 3 + 2] - coord[c * 3 + 2];
        cdiff_s[wv][lane * 3 + 0] = dx;
        cdiff_s[wv][lane * 3 + 1] = dy;
        cdiff_s[wv][lane * 3 + 2] = dz;
        radial_s[wv][lane] = dx * dx + dy * dy + dz * dz;
    }
    asm volatile("" ::: "memory");  // per-wave LDS is in-order (DScnt)

    const int row_m = rows_s[wv][m16];
    const int col_m = cols_s[wv][m16];
    const float* hr = h + (long)row_m * 64;
    const float* hc = h + (long)col_m * 64;
    const int abase = hi * 8;
    const int e_m = (int)e0 + m16;

    // ---- Layer e1: e_in[131->160] @ W_e1 -> [16 x 64] ----
    v8f acc1[4];
#pragma unroll
    for (int t = 0; t < 4; ++t)
#pragma unroll
        for (int j = 0; j < 8; ++j) acc1[t][j] = 0.0f;

#pragma unroll
    for (int c2 = 0; c2 < 2; ++c2) {
        v16h aR = a_frag(hr, c2 * 32 + abase);   // ks 0..63 of e_in
        v16h aC = a_frag(hc, c2 * 32 + abase);   // ks 64..127 of e_in
#pragma unroll
        for (int t = 0; t < 4; ++t) {
            const _Float16* wrow = &wt_e1[(t * 16 + m16) * WE1_STRIDE];
            acc1[t] = wmma_f16(aR, b_frag(wrow, c2 * 32 + hi * 16), acc1[t]);
            acc1[t] = wmma_f16(aC, b_frag(wrow, (c2 + 2) * 32 + hi * 16), acc1[t]);
        }
    }
    // Chunk 4: ks 128..159 = [radial, attr0, attr1, 0...]
    {
        v16h a4;
#pragma unroll
        for (int j = 0; j < 16; ++j) a4[j] = (_Float16)0.0f;
        float r0 = hi ? 0.0f : radial_s[wv][m16];
        float a1 = hi ? 0.0f : edge_attr[(long)e_m * 2 + 0];
        float a2 = hi ? 0.0f : edge_attr[(long)e_m * 2 + 1];
        a4[0] = (_Float16)r0; a4[1] = (_Float16)a1; a4[2] = (_Float16)a2;
#pragma unroll
        for (int t = 0; t < 4; ++t) {
            const _Float16* wrow = &wt_e1[(t * 16 + m16) * WE1_STRIDE];
            acc1[t] = wmma_f16(a4, b_frag(wrow, 128 + hi * 16), acc1[t]);
        }
    }
    // bias + SiLU, stash to LDS for next layer's A-fragments.
#pragma unroll
    for (int t = 0; t < 4; ++t) {
        int n = t * 16 + m16;
        float bias = b_e1[n];
#pragma unroll
        for (int r = 0; r < 8; ++r) {
            float x = silu(acc1[t][r] + bias);
            act[wv][(r + 8 * hi) * 68 + n] = x;
        }
    }
    asm volatile("" ::: "memory");

    // ---- Layer e2: [16x64] @ W_e2, SiLU -> edge_feat ----
    v8f feat[4];
#pragma unroll
    for (int t = 0; t < 4; ++t)
#pragma unroll
        for (int j = 0; j < 8; ++j) feat[t][j] = 0.0f;
    const float* arow = &act[wv][m16 * 68];
#pragma unroll
    for (int c2 = 0; c2 < 2; ++c2) {
        v16h a = a_frag(arow, c2 * 32 + abase);
#pragma unroll
        for (int t = 0; t < 4; ++t)
            feat[t] = wmma_f16(a, b_frag(&wt_e2[(t * 16 + m16) * W64_STRIDE],
                                         c2 * 32 + hi * 16), feat[t]);
    }
#pragma unroll
    for (int t = 0; t < 4; ++t) {
        int n = t * 16 + m16;
        float bias = b_e2[n];
#pragma unroll
        for (int r = 0; r < 8; ++r)
            feat[t][r] = silu(feat[t][r] + bias);
    }
    // edge_feat -> LDS (A operand of coord MLP); registers kept for scatter.
#pragma unroll
    for (int t = 0; t < 4; ++t) {
        int n = t * 16 + m16;
#pragma unroll
        for (int r = 0; r < 8; ++r)
            act[wv][(r + 8 * hi) * 68 + n] = feat[t][r];
    }
    asm volatile("" ::: "memory");

    // ---- coord MLP layer c1: edge_feat @ W_c1, SiLU ----
    v8f accc[4];
#pragma unroll
    for (int t = 0; t < 4; ++t)
#pragma unroll
        for (int j = 0; j < 8; ++j) accc[t][j] = 0.0f;
#pragma unroll
    for (int c2 = 0; c2 < 2; ++c2) {
        v16h a = a_frag(arow, c2 * 32 + abase);
#pragma unroll
        for (int t = 0; t < 4; ++t)
            accc[t] = wmma_f16(a, b_frag(&wt_c1[(t * 16 + m16) * W64_STRIDE],
                                         c2 * 32 + hi * 16), accc[t]);
    }
    // Need edge_feat for agg_h scatter AFTER c1 activations land in LDS, so
    // write c1 activations now.
#pragma unroll
    for (int t = 0; t < 4; ++t) {
        int n = t * 16 + m16;
        float bias = b_c1[n];
#pragma unroll
        for (int r = 0; r < 8; ++r)
            act[wv][(r + 8 * hi) * 68 + n] = silu(accc[t][r] + bias);
    }
    asm volatile("" ::: "memory");

    // ---- coord head (D->1) + trans scatter: lanes 0..15, one edge each ----
    if (lane < 16) {
        float s = 0.0f;
#pragma unroll 8
        for (int k = 0; k < 64; ++k)
            s += act[wv][lane * 68 + k] * W_c2[k];
        int r = rows_s[wv][lane];
        float tx = cdiff_s[wv][lane * 3 + 0] * s;
        float ty = cdiff_s[wv][lane * 3 + 1] * s;
        float tz = cdiff_s[wv][lane * 3 + 2] * s;
        tx = fminf(fmaxf(tx, -100.0f), 100.0f);
        ty = fminf(fmaxf(ty, -100.0f), 100.0f);
        tz = fminf(fmaxf(tz, -100.0f), 100.0f);
        atomicAdd(&agg_c[(long)r * 3 + 0], tx);
        atomicAdd(&agg_c[(long)r * 3 + 1], ty);
        atomicAdd(&agg_c[(long)r * 3 + 2], tz);
        atomicAdd(&cnt[r], 1.0f);
    }

    // ---- edge_feat scatter-add into agg_h (segment_sum by row) ----
#pragma unroll
    for (int t = 0; t < 4; ++t) {
        int n = t * 16 + m16;
#pragma unroll
        for (int r = 0; r < 8; ++r) {
            int m = r + 8 * hi;
            int rr = rows_s[wv][m];
            atomicAdd(&agg_h[(long)rr * 64 + n], feat[t][r]);
        }
    }
}

// ---------------- Node kernel: 8 waves/block, 16 nodes/wave ----------------
__global__ __launch_bounds__(256) void egcl_node_kernel(
    const float* __restrict__ h, const float* __restrict__ coord,
    const float* __restrict__ W_n1, const float* __restrict__ b_n1,
    const float* __restrict__ W_n2, const float* __restrict__ b_n2,
    const float* __restrict__ agg_h, const float* __restrict__ agg_c,
    const float* __restrict__ cnt, float* __restrict__ out, int Nn)
{
    __shared__ _Float16 wt_n1[64 * WN1_STRIDE];
    __shared__ _Float16 wt_n2[64 * W64_STRIDE];
    __shared__ float act[8][16 * 68];

    const int tid = threadIdx.x;
    for (int i = tid; i < 64 * 128; i += 256) {
        int n = i / 128, k = i % 128;
        wt_n1[n * WN1_STRIDE + k] = (_Float16)W_n1[k * 64 + n];
    }
    for (int i = tid; i < 64 * 64; i += 256) {
        int n = i / 64, k = i % 64;
        wt_n2[n * W64_STRIDE + k] = (_Float16)W_n2[k * 64 + n];
    }
    __syncthreads();

    const int wv = tid >> 5, lane = tid & 31;
    const int m16 = lane & 15, hi = lane >> 4;
    const int w = blockIdx.x * 8 + wv;
    const int n0 = w * 16;
    if (n0 >= Nn) return;

    const int nodeA = n0 + m16;
    const float* hrow = h + (long)nodeA * 64;
    const float* grow = agg_h + (long)nodeA * 64;
    const int abase = hi * 8;

    // ---- Layer n1: [h | agg_h] (K=128) @ W_n1, SiLU ----
    v8f acc1[4];
#pragma unroll
    for (int t = 0; t < 4; ++t)
#pragma unroll
        for (int j = 0; j < 8; ++j) acc1[t][j] = 0.0f;
#pragma unroll
    for (int c2 = 0; c2 < 2; ++c2) {
        v16h aH = a_frag(hrow, c2 * 32 + abase);   // ks 0..63
        v16h aG = a_frag(grow, c2 * 32 + abase);   // ks 64..127
#pragma unroll
        for (int t = 0; t < 4; ++t) {
            const _Float16* wrow = &wt_n1[(t * 16 + m16) * WN1_STRIDE];
            acc1[t] = wmma_f16(aH, b_frag(wrow, c2 * 32 + hi * 16), acc1[t]);
            acc1[t] = wmma_f16(aG, b_frag(wrow, (c2 + 2) * 32 + hi * 16), acc1[t]);
        }
    }
#pragma unroll
    for (int t = 0; t < 4; ++t) {
        int n = t * 16 + m16;
        float bias = b_n1[n];
#pragma unroll
        for (int r = 0; r < 8; ++r)
            act[wv][(r + 8 * hi) * 68 + n] = silu(acc1[t][r] + bias);
    }
    asm volatile("" ::: "memory");

    // ---- Layer n2 + residual ----
    v8f acc2[4];
#pragma unroll
    for (int t = 0; t < 4; ++t)
#pragma unroll
        for (int j = 0; j < 8; ++j) acc2[t][j] = 0.0f;
    const float* arow = &act[wv][m16 * 68];
#pragma unroll
    for (int c2 = 0; c2 < 2; ++c2) {
        v16h a = a_frag(arow, c2 * 32 + abase);
#pragma unroll
        for (int t = 0; t < 4; ++t)
            acc2[t] = wmma_f16(a, b_frag(&wt_n2[(t * 16 + m16) * W64_STRIDE],
                                         c2 * 32 + hi * 16), acc2[t]);
    }
#pragma unroll
    for (int t = 0; t < 4; ++t) {
        int n = t * 16 + m16;
        float bias = b_n2[n];
#pragma unroll
        for (int r = 0; r < 8; ++r) {
            int node = n0 + r + 8 * hi;
            out[(long)node * 64 + n] = h[(long)node * 64 + n] + acc2[t][r] + bias;
        }
    }

    // ---- coord_out = coord + agg_c / clip(cnt, 1) ----
    if (lane < 16) {
        int node = n0 + lane;
        float cv = cnt[node];
        cv = (cv < 1.0f) ? 1.0f : cv;
        float inv = 1.0f / cv;
        long cbase = (long)Nn * 64 + (long)node * 3;
#pragma unroll
        for (int j = 0; j < 3; ++j)
            out[cbase + j] = coord[(long)node * 3 + j] + agg_c[(long)node * 3 + j] * inv;
    }
}

extern "C" void kernel_launch(void* const* d_in, const int* in_sizes, int n_in,
                              void* d_out, int out_size, void* d_ws, size_t ws_size,
                              hipStream_t stream) {
    const float* h          = (const float*)d_in[0];
    const float* coord      = (const float*)d_in[1];
    const int*   edge_index = (const int*)d_in[2];
    const float* edge_attr  = (const float*)d_in[3];
    const float* W_e1 = (const float*)d_in[4];
    const float* b_e1 = (const float*)d_in[5];
    const float* W_e2 = (const float*)d_in[6];
    const float* b_e2 = (const float*)d_in[7];
    const float* W_n1 = (const float*)d_in[8];
    const float* b_n1 = (const float*)d_in[9];
    const float* W_n2 = (const float*)d_in[10];
    const float* b_n2 = (const float*)d_in[11];
    const float* W_c1 = (const float*)d_in[12];
    const float* b_c1 = (const float*)d_in[13];
    const float* W_c2 = (const float*)d_in[14];

    const int N = in_sizes[0] / 64;     // 50000
    const int E = in_sizes[3] / 2;      // 800000

    float* agg_h = (float*)d_ws;                       // [N,64]
    float* agg_c = agg_h + (size_t)N * 64;             // [N,3]
    float* cnt   = agg_c + (size_t)N * 3;              // [N]
    hipMemsetAsync(d_ws, 0, (size_t)N * 68 * sizeof(float), stream);

    // Edge pass: 4 waves/block * 16 edges/wave = 64 edges/block.
    int eblocks = (E + 63) / 64;
    egcl_edge_kernel<<<eblocks, 128, 0, stream>>>(
        h, coord, edge_index, edge_attr,
        W_e1, b_e1, W_e2, b_e2, W_c1, b_c1, W_c2,
        agg_h, agg_c, cnt, E);

    // Node pass: 8 waves/block * 16 nodes/wave = 128 nodes/block.
    int nblocks = (N + 127) / 128;
    egcl_node_kernel<<<nblocks, 256, 0, stream>>>(
        h, coord, W_n1, b_n1, W_n2, b_n2,
        agg_h, agg_c, cnt, (float*)d_out, N);
}
